// MSF_Module_47863115546697
// MI455X (gfx1250) — matrix operand order repaired and checked by reading, not compiled, not run on hardware
//
#include <hip/hip_runtime.h>

// ---------------------------------------------------------------------------
// Problem constants (from the reference)
// ---------------------------------------------------------------------------
#define NVOX   100000      // B*PER
#define NB     4           // batches
#define CSEM   64
#define CC     32
#define NCLS   26
#define K27    27
#define NTILES (NVOX/16)   // 6250, exact

typedef __attribute__((ext_vector_type(4)))  unsigned int uint4e;
typedef __attribute__((ext_vector_type(8)))  float        f32x8;
typedef __attribute__((ext_vector_type(16))) __bf16       bf16x16;

union AV {
    bf16x16 v;
    struct P { uint4e lo; uint4e hi; } p;
};

__device__ __forceinline__ unsigned short f2bf(float f) {
    unsigned x = __float_as_uint(f);
    unsigned r = x + 0x7fffu + ((x >> 16) & 1u);   // round-to-nearest-even
    return (unsigned short)(r >> 16);
}
__device__ __forceinline__ float sigm(float x) { return 1.f / (1.f + expf(-x)); }

// ---------------------------------------------------------------------------
// init: zero the small stats/coef region AND the zero-gather rows of the bf16
// activation buffers (row index NVOX). Runs every launch: graph-replay safe.
// ---------------------------------------------------------------------------
__global__ void k_init(float* sm, int n, unsigned short* xbZero, unsigned short* hbZero) {
    int i = blockIdx.x * blockDim.x + threadIdx.x;
    if (i < n) sm[i] = 0.f;
    if (i < CSEM) xbZero[i] = 0;
    if (i < CC)   hbZero[i] = 0;
}

// ---------------------------------------------------------------------------
// fusion gate: per-voxel sigmoid(feat_mos . w + b); accumulate batch sums of
// fm = gate*feat_sem (seg_mean numerator) and counts.
// ---------------------------------------------------------------------------
__global__ __launch_bounds__(256) void k_fuse(
    const float* __restrict__ fsem, const float* __restrict__ fmos,
    const float* __restrict__ spw, const float* __restrict__ spb,
    const int* __restrict__ bids, float* __restrict__ segsum, float* __restrict__ segcnt)
{
    __shared__ float w[CSEM];
    __shared__ float ls[NB * CSEM];
    __shared__ float lc[NB];
    int tid = threadIdx.x;
    if (tid < CSEM) w[tid] = spw[tid];
    if (tid < NB * CSEM) ls[tid] = 0.f;
    if (tid < NB) lc[tid] = 0.f;
    __syncthreads();

    int n = blockIdx.x * blockDim.x + tid;
    if (n < NVOX) {
        const float* mrow = fmos + (size_t)n * CSEM;
        float d = 0.f;
        #pragma unroll 8
        for (int c = 0; c < CSEM; ++c) d += mrow[c] * w[c];
        float s = sigm(d + spb[0]);
        int b = bids[n];
        const float* srow = fsem + (size_t)n * CSEM;
        atomicAdd(&lc[b], 1.f);
        for (int cc = 0; cc < CSEM; ++cc) {
            int c = (cc + tid) & (CSEM - 1);          // stagger to spread banks
            atomicAdd(&ls[b * CSEM + c], s * srow[c]);
        }
    }
    __syncthreads();
    if (tid < NB * CSEM) atomicAdd(&segsum[tid], ls[tid]);
    if (tid < NB)        atomicAdd(&segcnt[tid], lc[tid]);
}

// ---------------------------------------------------------------------------
// vec = softmax(seg_mean @ fus_ch_w + b) * 64   (4x64, one block)
// ---------------------------------------------------------------------------
__global__ void k_vec(const float* __restrict__ segsum, const float* __restrict__ segcnt,
                      const float* __restrict__ chw, const float* __restrict__ chb,
                      float* __restrict__ vec)
{
    __shared__ float mean[NB][CSEM];
    __shared__ float vv[NB][CSEM];
    __shared__ float mx[NB], sum[NB];
    int o = threadIdx.x, b = threadIdx.y;
    mean[b][o] = segsum[b * CSEM + o] / segcnt[b];
    __syncthreads();
    float a = chb[o];
    for (int c = 0; c < CSEM; ++c) a += mean[b][c] * chw[c * CSEM + o];
    vv[b][o] = a;
    __syncthreads();
    if (o == 0) {
        float m = -1e30f;
        for (int c = 0; c < CSEM; ++c) m = fmaxf(m, vv[b][c]);
        float s = 0.f;
        for (int c = 0; c < CSEM; ++c) s += expf(vv[b][c] - m);
        mx[b] = m; sum[b] = s;
    }
    __syncthreads();
    vec[b * CSEM + o] = expf(vv[b][o] - mx[b]) / sum[b] * (float)CSEM;
}

// ---------------------------------------------------------------------------
// feat = feat_sem * (1 + gate*vec[batch])  -> bf16 conv1 input
// ---------------------------------------------------------------------------
__global__ __launch_bounds__(256) void k_feat(
    const float* __restrict__ fsem, const float* __restrict__ fmos,
    const float* __restrict__ spw, const float* __restrict__ spb,
    const int* __restrict__ bids, const float* __restrict__ vec,
    unsigned short* __restrict__ XB)
{
    __shared__ float w[CSEM];
    int tid = threadIdx.x;
    if (tid < CSEM) w[tid] = spw[tid];
    __syncthreads();
    int n = blockIdx.x * blockDim.x + tid;
    if (n >= NVOX) return;
    const float* mrow = fmos + (size_t)n * CSEM;
    float d = 0.f;
    #pragma unroll 8
    for (int c = 0; c < CSEM; ++c) d += mrow[c] * w[c];
    float s = sigm(d + spb[0]);
    int b = bids[n];
    const float* srow = fsem + (size_t)n * CSEM;
    unsigned short* orow = XB + (size_t)n * CSEM;
    for (int c = 0; c < CSEM; ++c)
        orow[c] = f2bf(srow[c] * (1.f + s * vec[b * CSEM + c]));
}

// ---------------------------------------------------------------------------
// pack W[27][CIN][32] (f32) -> bf16 WMMA B-tiles.
// Tile (k,kt,nt) holds 512 bf16; slot [lane][i] = W[k][kt*32 + (lane>>4)*16 + i][nt*16 + (lane&15)]
// so each lane's 32B load is its exact 32x16 bf16 B fragment.
// ---------------------------------------------------------------------------
__global__ void k_pack(const float* __restrict__ W, unsigned short* __restrict__ out,
                       int KT, int CIN, int total)
{
    int s = blockIdx.x * blockDim.x + threadIdx.x;
    if (s >= total) return;
    int i    = s & 15;
    int lane = (s >> 4) & 31;
    int tile = s >> 9;
    int nt   = tile & 1;
    int ktk  = tile >> 1;
    int kt   = ktk % KT;
    int k    = ktk / KT;
    int cin  = kt * 32 + (lane >> 4) * 16 + i;
    int cout = nt * 16 + (lane & 15);
    out[s] = f2bf(W[((size_t)k * CIN + cin) * CC + cout]);
}

// ---------------------------------------------------------------------------
// Submanifold gather-GEMM conv via WMMA bf16 (f32 accum).
// One wave = one 16-voxel tile. A fragment in contiguous k order (same as k_pack B):
// lane (m = lane&15, half = lane>>4) -> channels [half*16..+15].
// D layout: element r -> row m = r + 8*half, col = lane&15 (+16 per n-tile).
// Missing neighbors gather from the all-zero row NVOX (branch-free k-loop, EXEC
// stays all-ones). Packed weights staged once per block into LDS.
// ---------------------------------------------------------------------------
template <int CIN, bool HAS_BIAS, bool HAS_BF, bool HAS_STATS>
__global__ __launch_bounds__(256) void k_conv(
    const unsigned short* __restrict__ Xin, const unsigned short* __restrict__ Wp,
    const int* __restrict__ nbr, const float* __restrict__ bias,
    float* __restrict__ outF, unsigned short* __restrict__ outB,
    float* __restrict__ stats)
{
    constexpr int KT = CIN / 32;
    constexpr int WELEMS = K27 * KT * 2 * 512;     // packed bf16 weight elements
    __shared__ unsigned short wlds[WELEMS];
    __shared__ float ls[CC], lq[CC];

    const int tid  = threadIdx.x;
    const int lane = tid & 31;
    const int wave = tid >> 5;
    const int tile = blockIdx.x * 8 + wave;
    const int m    = lane & 15;
    const int half = lane >> 4;
    const bool active = (tile < NTILES);
    const int vox = active ? (tile * 16 + m) : 0;

    // stage packed weights into LDS (one pass per block, reused by 8 waves x 27 taps)
    {
        const uint4e* src = (const uint4e*)Wp;
        uint4e*       dst = (uint4e*)wlds;
        for (int t = tid; t < WELEMS / 8; t += 256) dst[t] = src[t];
    }
    if (HAS_STATS && tid < CC) { ls[tid] = 0.f; lq[tid] = 0.f; }
    __syncthreads();

    int nb[K27];
    {
        const int* nrow = nbr + (size_t)vox * K27;
        #pragma unroll
        for (int k = 0; k < K27; ++k) nb[k] = active ? nrow[k] : -1;
    }

    f32x8 acc0, acc1;
    #pragma unroll
    for (int r = 0; r < 8; ++r) { acc0[r] = 0.f; acc1[r] = 0.f; }

    for (int k = 0; k < K27; ++k) {
        int g  = nb[k];
        int gi = (g >= 0) ? g : NVOX;              // zero row for absent neighbors
        if (k + 1 < K27) {                         // prefetch next tap's gather row
            int gn = nb[k + 1];
            __builtin_prefetch(Xin + (size_t)((gn >= 0) ? gn : NVOX) * CIN, 0, 3);
        }
        #pragma unroll
        for (int kt = 0; kt < KT; ++kt) {
            const unsigned short* xb = Xin + (size_t)gi * CIN + kt * 32 + half * 16;
            AV a;
            a.p.lo = *(const uint4e*)(xb);
            a.p.hi = *(const uint4e*)(xb + 8);
            const unsigned short* wb = wlds + (k * KT + kt) * 1024 + lane * 16;
            AV b0, b1;
            b0.p.lo = *(const uint4e*)(wb);
            b0.p.hi = *(const uint4e*)(wb + 8);
            b1.p.lo = *(const uint4e*)(wb + 512);
            b1.p.hi = *(const uint4e*)(wb + 520);
            acc0 = __builtin_amdgcn_wmma_f32_16x16x32_bf16(false, a.v, false, b0.v,
                                                           (short)0, acc0, false, false);
            acc1 = __builtin_amdgcn_wmma_f32_16x16x32_bf16(false, a.v, false, b1.v,
                                                           (short)0, acc1, false, false);
        }
    }

    float s0 = 0.f, q0 = 0.f, s1 = 0.f, q1 = 0.f;
    const int col = lane & 15;
    if (active) {
        float bi0 = HAS_BIAS ? bias[col]      : 0.f;
        float bi1 = HAS_BIAS ? bias[16 + col] : 0.f;
        #pragma unroll
        for (int r = 0; r < 8; ++r) {
            size_t row = (size_t)tile * 16 + r + 8 * half;
            float v0 = acc0[r] + bi0;
            float v1 = acc1[r] + bi1;
            outF[row * CC + col]      = v0;
            outF[row * CC + 16 + col] = v1;
            if (HAS_BF) {
                outB[row * CC + col]      = f2bf(v0);
                outB[row * CC + 16 + col] = f2bf(v1);
            }
            s0 += v0; q0 += v0 * v0; s1 += v1; q1 += v1 * v1;
        }
    }
    if (HAS_STATS) {
        atomicAdd(&ls[col], s0);      atomicAdd(&lq[col], q0);
        atomicAdd(&ls[16 + col], s1); atomicAdd(&lq[16 + col], q1);
        __syncthreads();
        if (tid < CC) {
            atomicAdd(&stats[tid],      ls[tid]);
            atomicAdd(&stats[CC + tid], lq[tid]);
        }
    }
}

// ---------------------------------------------------------------------------
// BN coefficients from (sum,sumsq): a = g*rsqrt(var+eps), b = beta - mean*a
// ---------------------------------------------------------------------------
__global__ void k_coef(const float* __restrict__ stats, const float* __restrict__ gamma,
                       const float* __restrict__ beta, float* __restrict__ coef)
{
    int c = threadIdx.x;
    if (c >= CC) return;
    float invN = 1.f / (float)NVOX;
    float mean = stats[c] * invN;
    float var  = stats[CC + c] * invN - mean * mean;
    float a = gamma[c] * rsqrtf(var + 1e-4f);
    coef[c]      = a;
    coef[CC + c] = beta[c] - mean * a;
}

// o = bf16(relu(bn(t)))
__global__ __launch_bounds__(256) void e_bn_relu(
    const float* __restrict__ T, const float* __restrict__ coef,
    unsigned short* __restrict__ outB)
{
    int i = blockIdx.x * blockDim.x + threadIdx.x;   // exactly NVOX*CC
    int c = i & (CC - 1);
    float v = fmaxf(fmaf(T[i], coef[c], coef[CC + c]), 0.f);
    outB[i] = f2bf(v);
}

// h = relu(bn(t)+h), in place; accumulate per-batch sum & max (attention)
__global__ __launch_bounds__(256) void e_res_att(
    const float* __restrict__ T, const float* __restrict__ coef,
    float* __restrict__ H, const int* __restrict__ bids,
    float* __restrict__ ssumG, float* __restrict__ smaxG)
{
    __shared__ float    ls[NB * CC];
    __shared__ unsigned lm[NB * CC];
    int tid = threadIdx.x;
    if (tid < NB * CC) { ls[tid] = 0.f; lm[tid] = 0u; }
    __syncthreads();
    int i = blockIdx.x * blockDim.x + tid;
    int c = i & (CC - 1);
    float v = fmaxf(fmaf(T[i], coef[c], coef[CC + c]) + H[i], 0.f);
    H[i] = v;
    int b = bids[i >> 5];
    atomicAdd(&ls[b * CC + c], v);
    atomicMax(&lm[b * CC + c], __float_as_uint(v));   // v >= 0 -> monotone bits
    __syncthreads();
    if (tid < NB * CC) {
        atomicAdd(&ssumG[tid], ls[tid]);
        atomicMax((unsigned*)&smaxG[tid], lm[tid]);
    }
}

// att = sigmoid(mlp(avg)+mlp(max)), one block (32 x 4)
__global__ void k_att(const float* __restrict__ ssum, const float* __restrict__ smax,
                      const float* __restrict__ cnt,
                      const float* __restrict__ W1, const float* __restrict__ W2,
                      float* __restrict__ att)
{
    __shared__ float avg[NB][CC], mxv[NB][CC];
    __shared__ float hA[NB][4], hM[NB][4];
    int c = threadIdx.x, b = threadIdx.y;
    avg[b][c] = ssum[b * CC + c] / cnt[b];
    mxv[b][c] = smax[b * CC + c];
    __syncthreads();
    if (c < 4) {
        float sa = 0.f, sm = 0.f;
        for (int k = 0; k < CC; ++k) {
            sa += avg[b][k] * W1[k * 4 + c];
            sm += mxv[b][k] * W1[k * 4 + c];
        }
        hA[b][c] = fmaxf(sa, 0.f);
        hM[b][c] = fmaxf(sm, 0.f);
    }
    __syncthreads();
    float s = 0.f;
    for (int j = 0; j < 4; ++j) s += (hA[b][j] + hM[b][j]) * W2[j * CC + c];
    att[b * CC + c] = sigm(s);
}

// h2 = h*att[batch] -> f32 residual copy + bf16 conv input
__global__ __launch_bounds__(256) void e_applyatt(
    const float* __restrict__ H, const float* __restrict__ att,
    const int* __restrict__ bids, float* __restrict__ H2,
    unsigned short* __restrict__ HB)
{
    int i = blockIdx.x * blockDim.x + threadIdx.x;
    int c = i & (CC - 1);
    float v = H[i] * att[bids[i >> 5] * CC + c];
    H2[i] = v;
    HB[i] = f2bf(v);
}

// h3 = relu(bn(t)+h2), store f32; accumulate stats for the final BN
__global__ __launch_bounds__(256) void e_res_stats(
    const float* __restrict__ T, const float* __restrict__ coef,
    const float* __restrict__ H2, float* __restrict__ Hout,
    float* __restrict__ stats)
{
    __shared__ float ls[CC], lq[CC];
    int tid = threadIdx.x;
    if (tid < CC) { ls[tid] = 0.f; lq[tid] = 0.f; }
    __syncthreads();
    int i = blockIdx.x * blockDim.x + tid;
    int c = i & (CC - 1);
    float v = fmaxf(fmaf(T[i], coef[c], coef[CC + c]) + H2[i], 0.f);
    Hout[i] = v;
    atomicAdd(&ls[c], v);
    atomicAdd(&lq[c], v * v);
    __syncthreads();
    if (tid < CC) {
        atomicAdd(&stats[tid],      ls[tid]);
        atomicAdd(&stats[CC + tid], lq[tid]);
    }
}

// out = relu(bn(h3)) @ lin_w + lin_b
__global__ __launch_bounds__(256) void k_final(
    const float* __restrict__ H, const float* __restrict__ coef,
    const float* __restrict__ lw, const float* __restrict__ lb,
    float* __restrict__ out)
{
    __shared__ float w[CC * NCLS];
    __shared__ float bshr[NCLS];
    int tid = threadIdx.x;
    for (int t = tid; t < CC * NCLS; t += blockDim.x) w[t] = lw[t];
    if (tid < NCLS) bshr[tid] = lb[tid];
    __syncthreads();
    int n = blockIdx.x * blockDim.x + tid;
    if (n >= NVOX) return;
    float x[CC];
    const float* hr = H + (size_t)n * CC;
    #pragma unroll
    for (int c = 0; c < CC; ++c)
        x[c] = fmaxf(fmaf(hr[c], coef[c], coef[CC + c]), 0.f);
    float* orow = out + (size_t)n * NCLS;
    for (int j = 0; j < NCLS; ++j) {
        float a = bshr[j];
        #pragma unroll
        for (int c = 0; c < CC; ++c) a += x[c] * w[c * NCLS + j];
        orow[j] = a;
    }
}

// ---------------------------------------------------------------------------
// Host launcher
// ---------------------------------------------------------------------------
extern "C" void kernel_launch(void* const* d_in, const int* in_sizes, int n_in,
                              void* d_out, int out_size, void* d_ws, size_t ws_size,
                              hipStream_t stream)
{
    (void)in_sizes; (void)n_in; (void)out_size; (void)ws_size;
    const float* fsem   = (const float*)d_in[0];
    const float* fmos   = (const float*)d_in[1];
    const float* spw    = (const float*)d_in[2];
    const float* spb    = (const float*)d_in[3];
    const float* chw    = (const float*)d_in[4];
    const float* chb    = (const float*)d_in[5];
    const float* conv_w = (const float*)d_in[6];
    const float* conv_b = (const float*)d_in[7];
    const float* r1c1_w = (const float*)d_in[8];
    const float* r1g1   = (const float*)d_in[9];
    const float* r1b1   = (const float*)d_in[10];
    const float* r1c2_w = (const float*)d_in[11];
    const float* r1g2   = (const float*)d_in[12];
    const float* r1b2   = (const float*)d_in[13];
    const float* ca1_w  = (const float*)d_in[14];
    const float* ca2_w  = (const float*)d_in[15];
    const float* r2c1_w = (const float*)d_in[16];
    const float* r2g1   = (const float*)d_in[17];
    const float* r2b1   = (const float*)d_in[18];
    const float* r2c2_w = (const float*)d_in[19];
    const float* r2g2   = (const float*)d_in[20];
    const float* r2b2   = (const float*)d_in[21];
    const float* out_g  = (const float*)d_in[22];
    const float* out_b  = (const float*)d_in[23];
    const float* lin_w  = (const float*)d_in[24];
    const float* lin_b  = (const float*)d_in[25];
    const int*   bids   = (const int*)d_in[26];
    const int*   nbr    = (const int*)d_in[27];
    float* outp = (float*)d_out;

    // -------- workspace carve-up (all offsets 256B aligned) --------
    char* ws = (char*)d_ws;
    size_t o = 0;
    auto carve = [&](size_t bytes) { size_t r = o; o = (o + bytes + 255) & ~(size_t)255; return r; };
    const size_t SMALLF = 4096;
    size_t oSMALL = carve(SMALLF * 4);
    size_t oXB    = carve((size_t)(NVOX + 1) * CSEM * 2);  // bf16 conv1 input (+zero row)
    size_t oHB    = carve((size_t)(NVOX + 1) * CC * 2);    // bf16 conv input (+zero row)
    size_t oT     = carve((size_t)NVOX * CC * 4);    // conv output f32
    size_t oH     = carve((size_t)NVOX * CC * 4);    // residual stream
    size_t oH2    = carve((size_t)NVOX * CC * 4);    // post-attention copy
    size_t oWP1   = carve((size_t)K27 * 2 * 2 * 512 * 2);
    size_t oWPa   = carve((size_t)K27 * 2 * 512 * 2);
    size_t oWPb   = carve((size_t)K27 * 2 * 512 * 2);
    size_t oWPc   = carve((size_t)K27 * 2 * 512 * 2);
    size_t oWPd   = carve((size_t)K27 * 2 * 512 * 2);

    float* sm = (float*)(ws + oSMALL);
    float* segsum  = sm + 0;     // 4*64
    float* segcnt  = sm + 256;   // 4
    float* vec     = sm + 288;   // 4*64
    float* stats0  = sm + 544;   // 5 stages x 64
    float* coef0   = sm + 864;   // 5 stages x 64
    float* segsumH = sm + 1184;  // 4*32
    float* segmaxH = sm + 1312;  // 4*32
    float* att     = sm + 1440;  // 4*32

    unsigned short* XB  = (unsigned short*)(ws + oXB);
    unsigned short* HB  = (unsigned short*)(ws + oHB);
    float* T  = (float*)(ws + oT);
    float* H  = (float*)(ws + oH);
    float* H2 = (float*)(ws + oH2);
    unsigned short* WP1 = (unsigned short*)(ws + oWP1);
    unsigned short* WPa = (unsigned short*)(ws + oWPa);
    unsigned short* WPb = (unsigned short*)(ws + oWPb);
    unsigned short* WPc = (unsigned short*)(ws + oWPc);
    unsigned short* WPd = (unsigned short*)(ws + oWPd);

    const int gVox  = (NVOX + 255) / 256;        // 391
    const int gElem = (NVOX * CC) / 256;         // 12500, exact
    const int gConv = (NTILES + 7) / 8;          // 782 (8 waves/block)

    // 1) zero stats region + zero-gather rows
    k_init<<<(SMALLF + 255) / 256, 256, 0, stream>>>(sm, (int)SMALLF,
        XB + (size_t)NVOX * CSEM, HB + (size_t)NVOX * CC);
    // 2) fusion gate + batch sums
    k_fuse<<<gVox, 256, 0, stream>>>(fsem, fmos, spw, spb, bids, segsum, segcnt);
    // 3) channel softmax vector
    k_vec<<<1, dim3(64, 4), 0, stream>>>(segsum, segcnt, chw, chb, vec);
    // 4) conv1 input (bf16)
    k_feat<<<gVox, 256, 0, stream>>>(fsem, fmos, spw, spb, bids, vec, XB);
    // 5) pack all conv weights into WMMA B-tile layout
    k_pack<<<(K27*2*2*512 + 255)/256, 256, 0, stream>>>(conv_w, WP1, 2, 64, K27*2*2*512);
    k_pack<<<(K27*2*512 + 255)/256, 256, 0, stream>>>(r1c1_w, WPa, 1, 32, K27*2*512);
    k_pack<<<(K27*2*512 + 255)/256, 256, 0, stream>>>(r1c2_w, WPb, 1, 32, K27*2*512);
    k_pack<<<(K27*2*512 + 255)/256, 256, 0, stream>>>(r2c1_w, WPc, 1, 32, K27*2*512);
    k_pack<<<(K27*2*512 + 255)/256, 256, 0, stream>>>(r2c2_w, WPd, 1, 32, K27*2*512);

    // 6) conv1: 64->32, +bias, emit f32 (residual) + bf16 (next conv input)
    k_conv<64, true, true, false><<<gConv, 256, 0, stream>>>(XB, WP1, nbr, conv_b, H, HB, nullptr);
    // 7) res1 conv1 + BN stats
    k_conv<32, false, false, true><<<gConv, 256, 0, stream>>>(HB, WPa, nbr, nullptr, T, nullptr, stats0 + 0);
    k_coef<<<1, 32, 0, stream>>>(stats0 + 0, r1g1, r1b1, coef0 + 0);
    e_bn_relu<<<gElem, 256, 0, stream>>>(T, coef0 + 0, HB);
    // res1 conv2 + BN stats
    k_conv<32, false, false, true><<<gConv, 256, 0, stream>>>(HB, WPb, nbr, nullptr, T, nullptr, stats0 + 64);
    k_coef<<<1, 32, 0, stream>>>(stats0 + 64, r1g2, r1b2, coef0 + 64);
    e_res_att<<<gElem, 256, 0, stream>>>(T, coef0 + 64, H, bids, segsumH, segmaxH);
    // channel attention
    k_att<<<1, dim3(32, 4), 0, stream>>>(segsumH, segmaxH, segcnt, ca1_w, ca2_w, att);
    e_applyatt<<<gElem, 256, 0, stream>>>(H, att, bids, H2, HB);
    // res2 conv1
    k_conv<32, false, false, true><<<gConv, 256, 0, stream>>>(HB, WPc, nbr, nullptr, T, nullptr, stats0 + 128);
    k_coef<<<1, 32, 0, stream>>>(stats0 + 128, r2g1, r2b1, coef0 + 128);
    e_bn_relu<<<gElem, 256, 0, stream>>>(T, coef0 + 128, HB);
    // res2 conv2
    k_conv<32, false, false, true><<<gConv, 256, 0, stream>>>(HB, WPd, nbr, nullptr, T, nullptr, stats0 + 192);
    k_coef<<<1, 32, 0, stream>>>(stats0 + 192, r2g2, r2b2, coef0 + 192);
    e_res_stats<<<gElem, 256, 0, stream>>>(T, coef0 + 192, H2, H, stats0 + 256);
    // final BN + linear head
    k_coef<<<1, 32, 0, stream>>>(stats0 + 256, out_g, out_b, coef0 + 256);
    k_final<<<gVox, 256, 0, stream>>>(H, coef0 + 256, lin_w, lin_b, outp);
}
